// FieldLabelClassify_43671227466244
// MI455X (gfx1250) — matrix operand order repaired and checked by reading, not compiled, not run on hardware
//
#include <hip/hip_runtime.h>
#include <hip/hip_bf16.h>
#include <cstdint>

// Problem constants (match the reference).
constexpr int NB = 16;     // batch
constexpr int NS = 2048;   // sequence
constexpr int NE = 128;    // entities
constexpr int NH = 768;    // hidden
constexpr int NC = 457;    // classes

typedef __attribute__((ext_vector_type(16))) __bf16 v16bf;
typedef __attribute__((ext_vector_type(8)))  __bf16 v8bf;
typedef __attribute__((ext_vector_type(8)))  float  v8f;

// ---------------------------------------------------------------------------
// Kernel 0: one-time W (C,H) f32 -> bf16 copy. 0.7 MB result stays L2-resident
// and removes all v_cvt from kernel 2's inner loop.
// ---------------------------------------------------------------------------
__global__ __launch_bounds__(256) void wcvt_kernel(
    const float* __restrict__ W, __bf16* __restrict__ Wb)
{
  const int i = (blockIdx.x * 256 + threadIdx.x) * 4;   // quad index
  if (i < NC * NH) {
    const float4 v = *(const float4*)(W + i);
    Wb[i + 0] = (__bf16)v.x;
    Wb[i + 1] = (__bf16)v.y;
    Wb[i + 2] = (__bf16)v.z;
    Wb[i + 3] = (__bf16)v.w;
  }
}

// ---------------------------------------------------------------------------
// Kernel 1: ent_hidden[b,e,h] = (sum_s elist[b,s,e] * hs[b,s,h]) / len[b,e]
// Batched GEMM (E x H, K = S) via v_wmma_f32_16x16x32_bf16, wave32 layouts.
// Workgroup = 4 waves = 16(e) x 64(h) output block; K-tile = 64:
//   - A 16x64 tile (0/1 ints, exact in bf16) staged via int4 loads -> LDS
//   - B 64x64 f32 block staged via coalesced b128 loads -> LDS, transposed
//     on the LDS read (per-lane ds_load_b32 + cvt, co-executes with WMMA)
//   - 2 WMMAs per barrier phase, next tile prefetched (global_prefetch_b8)
// ---------------------------------------------------------------------------
__global__ __launch_bounds__(128) void ent_pool_kernel(
    const float* __restrict__ hs,       // (B,S,H) f32
    const int*   __restrict__ elist,    // (B,S,E) int32 0/1
    const int*   __restrict__ elen,     // (B,E)   int32
    __bf16*      __restrict__ ent_ws)   // (B,E,H) bf16 out
{
  constexpr int KT = 64;
  __shared__ __bf16 a_lds[16][KT];      // [m][k] rows = 128B, 16B-aligned frags
  __shared__ float  b_lds[KT][68];      // [k][n], +4 pad floats per row

  const int lane = threadIdx.x & 31;
  const int wave = threadIdx.x >> 5;
  const int hi   = lane >> 4;           // lane half (0: lanes 0-15, 1: 16-31)
  const int mrow = lane & 15;

  constexpr int NGRP = NH / 64;         // 12 h-groups of 64
  constexpr int NET  = NE / 16;         // 8 e-tiles
  const int blk = blockIdx.x;
  const int b   = blk / (NET * NGRP);
  const int rem = blk % (NET * NGRP);
  const int e0  = (rem / NGRP) * 16;
  const int h0b = (rem % NGRP) * 64;    // block's h base

  // Per-lane K bases per ISA 16-bit A / B fragment layouts (wave32).
  const int akb  = hi ? 8 : 0;          // A: K in {akb..+7} u {akb+16..+23}
  const int bkb  = hi ? 8 : 0;          // B: K in {bkb..+7} u {bkb+16..+23}
  const int nloc = wave * 16 + mrow;    // column within the 64-wide block

  // Staging roles (all 128 threads).
  const int m4 = (threadIdx.x * 4) & 15;   // A: 4 consecutive entities
  const int kr = threadIdx.x >> 2;         // A/B: k row 0..31 (and +32)
  const int q  = threadIdx.x & 3;          // B: 16-float quarter of a row

  v8f acc = {};

  for (int k0 = 0; k0 < NS; k0 += KT) {
    // ---- Stage A: 16 x 64 ints -> bf16 LDS (int4 loads, contiguous in e).
#pragma unroll
    for (int half = 0; half < 2; ++half) {
      const int k = kr + half * 32;
      const int4 v = *(const int4*)&elist[((size_t)b * NS + k0 + k) * NE + e0 + m4];
      a_lds[m4 + 0][k] = (__bf16)(float)v.x;
      a_lds[m4 + 1][k] = (__bf16)(float)v.y;
      a_lds[m4 + 2][k] = (__bf16)(float)v.z;
      a_lds[m4 + 3][k] = (__bf16)(float)v.w;
    }
    // ---- Stage B: 64 x 64 f32 block, coalesced 16B loads/stores.
#pragma unroll
    for (int half = 0; half < 2; ++half) {
      const int k = kr + half * 32;
      const float* src = &hs[((size_t)b * NS + k0 + k) * NH + h0b + q * 16];
#pragma unroll
      for (int j = 0; j < 4; ++j)
        *(float4*)&b_lds[k][q * 16 + j * 4] = ((const float4*)src)[j];
    }
    // ---- Prefetch next K-tile toward this WGP (global_prefetch_b8).
    if (k0 + KT < NS) {
      __builtin_prefetch(&hs[((size_t)b * NS + k0 + KT + kr) * NH + h0b + q * 16], 0, 1);
      __builtin_prefetch(&elist[((size_t)b * NS + k0 + KT + kr) * NE + e0 + m4], 0, 1);
    }
    __syncthreads();

    // ---- Two 16x16x32 WMMAs over the 64-deep tile.
#pragma unroll
    for (int kk = 0; kk < KT; kk += 32) {
      const v8bf alo = *(const v8bf*)&a_lds[mrow][kk + akb];
      const v8bf ahi = *(const v8bf*)&a_lds[mrow][kk + akb + 16];
      v16bf a;
#pragma unroll
      for (int i = 0; i < 8; ++i) { a[i] = alo[i]; a[i + 8] = ahi[i]; }

      v16bf bb;                          // transpose on LDS read + cvt
#pragma unroll
      for (int i = 0; i < 16; ++i)
        bb[i] = (__bf16)b_lds[kk + bkb + (i & 7) + ((i >> 3) << 4)][nloc];

      acc = __builtin_amdgcn_wmma_f32_16x16x32_bf16(
          false, a, false, bb, (short)0, acc, false, false);
    }
    __syncthreads();
  }

  // Store D tile (M = r + 8*hi, N = lane&15), folding in 1/entities_len.
#pragma unroll
  for (int r = 0; r < 8; ++r) {
    const int e = e0 + r + (hi ? 8 : 0);
    const float len = (float)elen[b * NE + e];
    ent_ws[((size_t)b * NE + e) * NH + h0b + nloc] = (__bf16)(acc[r] / len);
  }
}

// ---------------------------------------------------------------------------
// Kernel 2: logits[b,e,c] = sum_h ent[b,e,h] * W[c,h] + bias[c]
// One wave per 16x16 (e,c) tile; K = H = 768 (24 wmma steps). Both operands
// now bf16 and row-contiguous -> pure 16B vector loads + WMMA, no cvt.
// ---------------------------------------------------------------------------
__global__ __launch_bounds__(32) void logits_kernel(
    const __bf16* __restrict__ ent_ws,  // (B,E,H) bf16
    const __bf16* __restrict__ Wb,      // (C,H)   bf16
    const float*  __restrict__ bias,    // (C,)    f32
    float*        __restrict__ logits)  // (B,E,C) f32 out
{
  const int lane = threadIdx.x & 31;
  const int hi   = lane >> 4;
  const int mrow = lane & 15;

  constexpr int CT = (NC + 15) / 16;    // 29 c-tiles (last partial)
  const int blk = blockIdx.x;
  const int b   = blk / (8 * CT);
  const int rem = blk % (8 * CT);
  const int e0  = (rem / CT) * 16;
  const int c0  = (rem % CT) * 16;

  const int akb = hi ? 8 : 0;
  const int bkb = hi ? 8 : 0;
  const int c   = c0 + mrow;
  const int c_eff = (c < NC) ? c : (NC - 1);   // clamp; padded cols not stored

  const __bf16* arow = ent_ws + ((size_t)b * NE + e0 + mrow) * NH;
  const __bf16* wrow = Wb + (size_t)c_eff * NH;

  v8f acc = {};
  for (int k0 = 0; k0 < NH; k0 += 32) {
    const v8bf alo = *(const v8bf*)(arow + k0 + akb);
    const v8bf ahi = *(const v8bf*)(arow + k0 + akb + 16);
    const v8bf blo = *(const v8bf*)(wrow + k0 + bkb);
    const v8bf bhi = *(const v8bf*)(wrow + k0 + bkb + 16);
    v16bf a, bb;
#pragma unroll
    for (int i = 0; i < 8; ++i) {
      a[i] = alo[i];  a[i + 8] = ahi[i];
      bb[i] = blo[i]; bb[i + 8] = bhi[i];
    }
    acc = __builtin_amdgcn_wmma_f32_16x16x32_bf16(
        false, a, false, bb, (short)0, acc, false, false);
  }

#pragma unroll
  for (int r = 0; r < 8; ++r) {
    const int M  = r + (hi ? 8 : 0);
    const int cc = c0 + mrow;
    if (cc < NC)
      logits[((size_t)b * NE + e0 + M) * NC + cc] = acc[r] + bias[cc];
  }
}

// ---------------------------------------------------------------------------
// Kernel 3: per-row masked NLL. One wave per (b,e) row of 457 logits;
// registers cache the row (15 vals/lane), shfl_xor tree reductions.
// ---------------------------------------------------------------------------
__global__ __launch_bounds__(256) void nll_kernel(
    const float* __restrict__ logits,   // (B*E, C)
    const int*   __restrict__ attr,     // (B*E)
    const unsigned char* __restrict__ mask, // (B*E) bool
    float* __restrict__ row_nll,        // (B*E)
    float* __restrict__ row_m)          // (B*E)
{
  const int lane = threadIdx.x & 31;
  const int wave = threadIdx.x >> 5;
  const int row  = blockIdx.x * 8 + wave;       // < B*E = 2048
  const float* lr = logits + (size_t)row * NC;

  float vals[15];
  float mx = -__builtin_inff();
#pragma unroll
  for (int i = 0; i < 15; ++i) {
    const int cc = lane + i * 32;
    const float v = (cc < NC) ? lr[cc] : -__builtin_inff();
    vals[i] = v;
    mx = fmaxf(mx, v);
  }
#pragma unroll
  for (int off = 16; off > 0; off >>= 1)
    mx = fmaxf(mx, __shfl_xor(mx, off, 32));

  float s = 0.f;
#pragma unroll
  for (int i = 0; i < 15; ++i) {
    const int cc = lane + i * 32;
    if (cc < NC) s += __expf(vals[i] - mx);
  }
#pragma unroll
  for (int off = 16; off > 0; off >>= 1)
    s += __shfl_xor(s, off, 32);

  if (lane == 0) {
    const float lse = mx + __logf(s);
    const float tgt = lr[attr[row]];
    const float m   = mask[row] ? 1.f : 0.f;
    row_nll[row] = (lse - tgt) * m;
    row_m[row]   = m;
  }
}

// ---------------------------------------------------------------------------
// Kernel 4: deterministic fixed-tree reduction of 2048 rows -> scalar loss.
// ---------------------------------------------------------------------------
__global__ __launch_bounds__(256) void reduce_kernel(
    const float* __restrict__ row_nll,
    const float* __restrict__ row_m,
    float* __restrict__ out)
{
  __shared__ float s1[256], s2[256];
  float a = 0.f, m = 0.f;
  for (int i = threadIdx.x; i < NB * NE; i += 256) { a += row_nll[i]; m += row_m[i]; }
  s1[threadIdx.x] = a; s2[threadIdx.x] = m;
  __syncthreads();
  for (int off = 128; off > 0; off >>= 1) {
    if ((int)threadIdx.x < off) {
      s1[threadIdx.x] += s1[threadIdx.x + off];
      s2[threadIdx.x] += s2[threadIdx.x + off];
    }
    __syncthreads();
  }
  if (threadIdx.x == 0) out[0] = s1[0] / s2[0];
}

// ---------------------------------------------------------------------------
// Host launcher. Input order: hidden_states, field_labels, label_masks,
// entities_attr, entities_list, entities_len, entities_mask, W, b.
// (field_labels / label_masks are unused by the reference output.)
// ---------------------------------------------------------------------------
extern "C" void kernel_launch(void* const* d_in, const int* in_sizes, int n_in,
                              void* d_out, int out_size, void* d_ws, size_t ws_size,
                              hipStream_t stream) {
  const float* hs    = (const float*)d_in[0];
  const int*   attr  = (const int*)d_in[3];
  const int*   elist = (const int*)d_in[4];
  const int*   elen  = (const int*)d_in[5];
  const unsigned char* emask = (const unsigned char*)d_in[6]; // bool array
  const float* W     = (const float*)d_in[7];
  const float* bias  = (const float*)d_in[8];
  float* out = (float*)d_out;

  // Workspace layout (16B+ aligned slabs).
  char* ws = (char*)d_ws;
  __bf16* ent_ws = (__bf16*)ws;                                  // 3,145,728 B
  size_t  off    = (size_t)NB * NE * NH * 2;
  float*  logits = (float*)(ws + off);                           // 3,743,744 B
  off += (size_t)NB * NE * NC * 4;
  __bf16* Wb     = (__bf16*)(ws + off);                          //   701,952 B
  off += (size_t)NC * NH * 2;
  float*  row_nll = (float*)(ws + off);                          //     8,192 B
  off += (size_t)NB * NE * 4;
  float*  row_m   = (float*)(ws + off);                          //     8,192 B

  constexpr int NGRP = NH / 64;        // 12
  constexpr int CT   = (NC + 15) / 16; // 29

  wcvt_kernel<<<(NC * NH / 4 + 255) / 256, 256, 0, stream>>>(W, Wb);
  ent_pool_kernel<<<NB * (NE / 16) * NGRP, 128, 0, stream>>>(hs, elist, elen, ent_ws);
  logits_kernel<<<NB * (NE / 16) * CT, 32, 0, stream>>>(ent_ws, Wb, bias, logits);
  nll_kernel<<<(NB * NE) / 8, 256, 0, stream>>>(logits, attr, emask, row_nll, row_m);
  reduce_kernel<<<1, 256, 0, stream>>>(row_nll, row_m, out);
}
